// MolGraphSAGE_72645076844772
// MI455X (gfx1250) — compile-verified, matrix operand order checked by reading, not run on hardware
//
#include <hip/hip_runtime.h>
#include <hip/hip_bf16.h>

#define N_NODES   50000
#define N_PAD     50048      // 64-row aligned (782 * 64)
#define N_GRAPHS  2000
#define BN_EPS    1e-5f

typedef __attribute__((ext_vector_type(16))) __bf16        v16bf;
typedef __attribute__((ext_vector_type(8)))  float         v8f;
typedef __attribute__((ext_vector_type(4)))  unsigned int  v4u;

union bfrag { v16bf v; v4u u[2]; };

// K index inside a bf16 16x32 A (or 32x16 B) fragment, per CDNA5 ISA layout:
// lanes 0-15 (half=0): VGPR0-3 -> K=0..7, VGPR4-7 -> K=16..23
// lanes 16-31 (half=1): VGPR0-3 -> K=8..15, VGPR4-7 -> K=24..31
__device__ __forceinline__ int wmma_kmap(int e, int half) {
  return (e < 8) ? (half * 8 + e) : (8 + half * 8 + e);
}

// ---------------- graph aggregation ----------------

__global__ void k_edge_deg(const int* __restrict__ dst, int E, float* __restrict__ deg) {
  for (int e = blockIdx.x * blockDim.x + threadIdx.x; e < E; e += gridDim.x * blockDim.x)
    atomicAdd(&deg[dst[e]], 1.0f);
}

__global__ void k_inv_deg(float* __restrict__ deg, int n) {
  int i = blockIdx.x * blockDim.x + threadIdx.x;
  if (i < n) deg[i] = 1.0f / fmaxf(deg[i], 1.0f);
}

// generic (used for d=38)
__global__ void k_scatter(const int* __restrict__ src, const int* __restrict__ dst,
                          const float* __restrict__ x, float* __restrict__ agg,
                          int E, int d) {
  size_t total = (size_t)E * d;
  for (size_t i = (size_t)blockIdx.x * blockDim.x + threadIdx.x; i < total;
       i += (size_t)gridDim.x * blockDim.x) {
    int e = (int)(i / d);
    int f = (int)(i - (size_t)e * d);
    atomicAdd(&agg[(size_t)dst[e] * d + f], x[(size_t)src[e] * d + f]);
  }
}

// specialized d=128 (shift/mask indexing, L2-resident atomics)
__global__ void k_scatter128(const int* __restrict__ src, const int* __restrict__ dst,
                             const float* __restrict__ x, float* __restrict__ agg,
                             int E) {
  size_t total = (size_t)E << 7;
  for (size_t i = (size_t)blockIdx.x * blockDim.x + threadIdx.x; i < total;
       i += (size_t)gridDim.x * blockDim.x) {
    int e = (int)(i >> 7);
    int f = (int)(i & 127);
    atomicAdd(&agg[((size_t)dst[e] << 7) + f], x[((size_t)src[e] << 7) + f]);
  }
}

// ---------------- bf16 conversion / packing ----------------

// agg[n,d] * invdeg[row] -> bf16 [*,dpad] row-major (zero K-pad)
__global__ void k_cvt_agg(const float* __restrict__ agg, const float* __restrict__ invdeg,
                          __bf16* __restrict__ out, int n, int d, int dpad) {
  size_t total = (size_t)n * dpad;
  for (size_t i = (size_t)blockIdx.x * blockDim.x + threadIdx.x; i < total;
       i += (size_t)gridDim.x * blockDim.x) {
    int r = (int)(i / dpad);
    int c = (int)(i - (size_t)r * dpad);
    float v = (c < d) ? agg[(size_t)r * d + c] * invdeg[r] : 0.0f;
    out[i] = (__bf16)v;
  }
}

// x[n,d] -> bf16 [*,dpad] row-major (zero K-pad)
__global__ void k_cvt_rows(const float* __restrict__ x, __bf16* __restrict__ out,
                           int n, int d, int dpad) {
  size_t total = (size_t)n * dpad;
  for (size_t i = (size_t)blockIdx.x * blockDim.x + threadIdx.x; i < total;
       i += (size_t)gridDim.x * blockDim.x) {
    int r = (int)(i / dpad);
    int c = (int)(i - (size_t)r * dpad);
    float v = (c < d) ? x[(size_t)r * d + c] : 0.0f;
    out[i] = (__bf16)v;
  }
}

// W[K,128] f32 -> bf16 packed in B-fragment order: [kblock][coltile][lane][e],
// so each lane's 16-element fragment is 32 contiguous bytes.
__global__ void k_pack_w(const float* __restrict__ W, __bf16* __restrict__ Wp,
                         int K, int Kpad) {
  int total = (Kpad >> 5) * 8 * 32 * 16;
  for (int i = blockIdx.x * blockDim.x + threadIdx.x; i < total;
       i += gridDim.x * blockDim.x) {
    int e    = i & 15;
    int lane = (i >> 4) & 31;
    int ct   = (i >> 9) & 7;
    int kb   = i >> 12;
    int half = lane >> 4;
    int k    = kb * 32 + wmma_kmap(e, half);
    int col  = ct * 16 + (lane & 15);
    Wp[i] = (k < K) ? (__bf16)W[(size_t)k * 128 + col] : (__bf16)0.0f;
  }
}

// ---------------- WMMA GEMM: out = Aa@Wl + Ah@Wr + bias ----------------
// Aa, Ah: [N_PAD, K] bf16 row-major; Wpl, Wpr: packed B fragments;
// out: [N_PAD, 128] f32. Grid: N_PAD/64 blocks of 256 threads (8 waves).
// Wave w handles columns [16w,16w+16) for a 64-row strip (4 row tiles).
// K is a compile-time constant so the fully-unrolled loop uses only
// immediate address offsets; all fragments of a k-block are loaded before
// the 8 WMMAs so loads stay in flight under partial s_wait_loadcnt.
template <int K>
__global__ void k_gemm_sage(const __bf16* __restrict__ Aa, const __bf16* __restrict__ Ah,
                            const __bf16* __restrict__ Wpl, const __bf16* __restrict__ Wpr,
                            const float* __restrict__ bias, float* __restrict__ out) {
  const int lane    = threadIdx.x & 31;
  const int wave    = threadIdx.x >> 5;    // col tile 0..7
  const int rowBase = blockIdx.x * 64;
  const int m       = lane & 15;
  const int half    = lane >> 4;
  const int colBase = wave * 16;

  v8f acc[4];
  const float bv = bias[colBase + m];
#pragma unroll
  for (int t = 0; t < 4; ++t)
#pragma unroll
    for (int r = 0; r < 8; ++r) acc[t][r] = bv;

  // fixed per-lane bases; everything else is an immediate offset
  const __bf16* aBase  = Aa + (size_t)(rowBase + m) * K + 8 * half;
  const __bf16* hBase  = Ah + (size_t)(rowBase + m) * K + 8 * half;
  const __bf16* blBase = Wpl + ((wave * 32 + lane) << 4);
  const __bf16* brBase = Wpr + ((wave * 32 + lane) << 4);

  constexpr int NKB = K / 32;
#pragma unroll
  for (int kb = 0; kb < NKB; ++kb) {
    bfrag fbl, fbr, fa[4], fh[4];
    const int bOff = kb * 4096;            // packed-B elements per k-block
    fbl.u[0] = *(const v4u*)(blBase + bOff);
    fbl.u[1] = *(const v4u*)(blBase + bOff + 8);
    fbr.u[0] = *(const v4u*)(brBase + bOff);
    fbr.u[1] = *(const v4u*)(brBase + bOff + 8);
#pragma unroll
    for (int t = 0; t < 4; ++t) {
      const int aOff = t * 16 * K + kb * 32;
      fa[t].u[0] = *(const v4u*)(aBase + aOff);
      fa[t].u[1] = *(const v4u*)(aBase + aOff + 16);
      fh[t].u[0] = *(const v4u*)(hBase + aOff);
      fh[t].u[1] = *(const v4u*)(hBase + aOff + 16);
    }
    // group WMMAs so each accumulator's D->C reuse is 4 instructions apart
#pragma unroll
    for (int t = 0; t < 4; ++t)
      acc[t] = __builtin_amdgcn_wmma_f32_16x16x32_bf16(false, fa[t].v, false, fbl.v,
                                                       (short)0, acc[t], false, false);
#pragma unroll
    for (int t = 0; t < 4; ++t)
      acc[t] = __builtin_amdgcn_wmma_f32_16x16x32_bf16(false, fh[t].v, false, fbr.v,
                                                       (short)0, acc[t], false, false);
  }

  float* op = out + ((size_t)rowBase + half * 8) * 128 + colBase + m;
#pragma unroll
  for (int t = 0; t < 4; ++t)
#pragma unroll
    for (int r = 0; r < 8; ++r)
      op[(size_t)(t * 16 + r) * 128] = acc[t][r];
}

// ---------------- BatchNorm statistics + BN/ReLU ----------------

__global__ void k_col_stats(const float* __restrict__ h, float* __restrict__ mean,
                            float* __restrict__ var, int n) {
  __shared__ float sh[256], sh2[256];
  int c = blockIdx.x, t = threadIdx.x;
  float s = 0.0f, s2 = 0.0f;
  for (int i = t; i < n; i += blockDim.x) {
    float v = h[(size_t)i * 128 + c];
    s += v; s2 += v * v;
  }
  sh[t] = s; sh2[t] = s2;
  __syncthreads();
  for (int w = 128; w > 0; w >>= 1) {
    if (t < w) { sh[t] += sh[t + w]; sh2[t] += sh2[t + w]; }
    __syncthreads();
  }
  if (t == 0) {
    float mm = sh[0] / n;
    mean[c] = mm;
    var[c]  = sh2[0] / n - mm * mm;   // biased variance (training-mode BN)
  }
}

__global__ void k_bn_relu(const float* __restrict__ hp, const float* __restrict__ mean,
                          const float* __restrict__ var, const float* __restrict__ g,
                          const float* __restrict__ be, float* __restrict__ h,
                          __bf16* __restrict__ hb, int n) {
  size_t total = (size_t)n * 128;
  for (size_t i = (size_t)blockIdx.x * blockDim.x + threadIdx.x; i < total;
       i += (size_t)gridDim.x * blockDim.x) {
    int c = (int)(i & 127);
    float z = (hp[i] - mean[c]) * rsqrtf(var[c] + BN_EPS) * g[c] + be[c];
    z = fmaxf(z, 0.0f);
    h[i]  = z;
    hb[i] = (__bf16)z;
  }
}

// ---------------- pooling + readout ----------------

__global__ void k_pool(const float* __restrict__ h, const int* __restrict__ batch,
                       float* __restrict__ gsum, float* __restrict__ gcnt, int n) {
  size_t total = (size_t)n * 128;
  for (size_t i = (size_t)blockIdx.x * blockDim.x + threadIdx.x; i < total;
       i += (size_t)gridDim.x * blockDim.x) {
    int node = (int)(i >> 7);
    int c    = (int)(i & 127);
    int gidx = batch[node];
    atomicAdd(&gsum[((size_t)gidx << 7) + c], h[i]);
    if (c == 0) atomicAdd(&gcnt[gidx], 1.0f);
  }
}

__global__ void k_readout1(const float* __restrict__ gsum, const float* __restrict__ gcnt,
                           const float* __restrict__ W1, const float* __restrict__ b1,
                           float* __restrict__ act) {
  int i = blockIdx.x * blockDim.x + threadIdx.x;   // over N_GRAPHS*64
  if (i >= N_GRAPHS * 64) return;
  int gidx = i >> 6, j = i & 63;
  float inv = 1.0f / fmaxf(gcnt[gidx], 1.0f);
  float s = b1[j];
  const float* gp = gsum + ((size_t)gidx << 7);
#pragma unroll 4
  for (int k = 0; k < 128; ++k) s += gp[k] * inv * W1[k * 64 + j];
  act[i] = fmaxf(s, 0.0f);
}

__global__ void k_readout2(const float* __restrict__ act, const float* __restrict__ W2,
                           const float* __restrict__ b2, float* __restrict__ out) {
  int gidx = blockIdx.x * blockDim.x + threadIdx.x;
  if (gidx >= N_GRAPHS) return;
  float s = b2[0];
  const float* ap = act + ((size_t)gidx << 6);
#pragma unroll 4
  for (int k = 0; k < 64; ++k) s += ap[k] * W2[k];
  out[gidx] = s;
}

// ---------------- host-side orchestration ----------------

extern "C" void kernel_launch(void* const* d_in, const int* in_sizes, int n_in,
                              void* d_out, int out_size, void* d_ws, size_t ws_size,
                              hipStream_t stream) {
  (void)n_in; (void)out_size; (void)ws_size;
  const float* x     = (const float*)d_in[0];
  const int*   ei    = (const int*)d_in[1];
  const int*   batch = (const int*)d_in[2];
  const float* Wl0   = (const float*)d_in[3];
  const float* Wr0   = (const float*)d_in[4];
  const float* b0    = (const float*)d_in[5];
  const float* WlA   = (const float*)d_in[6];  // [2,128,128]
  const float* WrA   = (const float*)d_in[7];
  const float* bA    = (const float*)d_in[8];  // [2,128]
  const float* gamma = (const float*)d_in[9];  // [3,128]
  const float* beta  = (const float*)d_in[10];
  const float* rW1   = (const float*)d_in[11];
  const float* rb1   = (const float*)d_in[12];
  const float* rW2   = (const float*)d_in[13];
  const float* rb2   = (const float*)d_in[14];
  float* out = (float*)d_out;

  const int E = in_sizes[1] / 2;
  const int* src = ei;
  const int* dst = ei + E;

  char* ws = (char*)d_ws;
  size_t off = 0;
  auto alloc = [&](size_t bytes) -> void* {
    void* p = ws + off;
    off += (bytes + 255) & ~(size_t)255;
    return p;
  };
  float*  deg  = (float*)alloc((size_t)N_NODES * 4);
  float*  agg  = (float*)alloc((size_t)N_NODES * 128 * 4);
  float*  h    = (float*)alloc((size_t)N_NODES * 128 * 4);
  float*  hpre = (float*)alloc((size_t)N_PAD * 128 * 4);
  __bf16* aggb = (__bf16*)alloc((size_t)N_PAD * 128 * 2);
  __bf16* hb   = (__bf16*)alloc((size_t)N_PAD * 128 * 2);
  __bf16* wlb  = (__bf16*)alloc(128 * 128 * 2);
  __bf16* wrb  = (__bf16*)alloc(128 * 128 * 2);
  float*  mean = (float*)alloc(128 * 4);
  float*  var  = (float*)alloc(128 * 4);
  float*  gsum = (float*)alloc((size_t)N_GRAPHS * 128 * 4);
  float*  gcnt = (float*)alloc((size_t)N_GRAPHS * 4);
  float*  gact = (float*)alloc((size_t)N_GRAPHS * 64 * 4);

  const int T = 256;
  const int gemmBlocks = N_PAD / 64;   // 782

  // in-degrees (shared by all layers)
  hipMemsetAsync(deg, 0, (size_t)N_NODES * 4, stream);
  k_edge_deg<<<2048, T, 0, stream>>>(dst, E, deg);
  k_inv_deg<<<(N_NODES + T - 1) / T, T, 0, stream>>>(deg, N_NODES);

  // ---- layer 0: 38 -> 128 (pad K to 64) ----
  hipMemsetAsync(agg, 0, (size_t)N_NODES * 38 * 4, stream);
  k_scatter<<<4096, T, 0, stream>>>(src, dst, x, agg, E, 38);
  k_cvt_agg<<<4096, T, 0, stream>>>(agg, deg, aggb, N_NODES, 38, 64);
  k_cvt_rows<<<4096, T, 0, stream>>>(x, hb, N_NODES, 38, 64);
  k_pack_w<<<32, T, 0, stream>>>(Wl0, wlb, 38, 64);
  k_pack_w<<<32, T, 0, stream>>>(Wr0, wrb, 38, 64);
  k_gemm_sage<64><<<gemmBlocks, T, 0, stream>>>(aggb, hb, wlb, wrb, b0, hpre);
  k_col_stats<<<128, T, 0, stream>>>(hpre, mean, var, N_NODES);
  k_bn_relu<<<4096, T, 0, stream>>>(hpre, mean, var, gamma, beta, h, hb, N_NODES);

  // ---- layers 1..2: 128 -> 128 ----
  for (int l = 0; l < 2; ++l) {
    hipMemsetAsync(agg, 0, (size_t)N_NODES * 128 * 4, stream);
    k_scatter128<<<4096, T, 0, stream>>>(src, dst, h, agg, E);
    k_cvt_agg<<<4096, T, 0, stream>>>(agg, deg, aggb, N_NODES, 128, 128);
    k_pack_w<<<64, T, 0, stream>>>(WlA + (size_t)l * 128 * 128, wlb, 128, 128);
    k_pack_w<<<64, T, 0, stream>>>(WrA + (size_t)l * 128 * 128, wrb, 128, 128);
    k_gemm_sage<128><<<gemmBlocks, T, 0, stream>>>(aggb, hb, wlb, wrb,
                                                   bA + (size_t)l * 128, hpre);
    k_col_stats<<<128, T, 0, stream>>>(hpre, mean, var, N_NODES);
    k_bn_relu<<<4096, T, 0, stream>>>(hpre, mean, var,
                                      gamma + (size_t)(l + 1) * 128,
                                      beta + (size_t)(l + 1) * 128, h, hb, N_NODES);
  }

  // ---- global mean pool + readout MLP ----
  hipMemsetAsync(gsum, 0, (size_t)N_GRAPHS * 128 * 4, stream);
  hipMemsetAsync(gcnt, 0, (size_t)N_GRAPHS * 4, stream);
  k_pool<<<4096, T, 0, stream>>>(h, batch, gsum, gcnt, N_NODES);
  k_readout1<<<(N_GRAPHS * 64 + T - 1) / T, T, 0, stream>>>(gsum, gcnt, rW1, rb1, gact);
  k_readout2<<<(N_GRAPHS + T - 1) / T, T, 0, stream>>>(gact, rW2, rb2, out);
}
